// hbPass_76390288326918
// MI455X (gfx1250) — compile-verified, matrix operand order checked by reading, not run on hardware
//
#include <hip/hip_runtime.h>

typedef __attribute__((ext_vector_type(16))) _Float16 v16h;
typedef __attribute__((ext_vector_type(8)))  _Float16 h8;
typedef __attribute__((ext_vector_type(8)))  float    v8f;
typedef __attribute__((ext_vector_type(4)))  float    f4;
typedef __attribute__((ext_vector_type(4)))  int      i4;

#define HH 56
#define WW 56
#define HP 58                // padded spatial (1-halo each side)
#define CIN 256
#define COUT 256
#define NB 32
#define HW (HH*WW)           // 3136
#define SPATIAL (NB*HW)      // 100352
#define STILE 256
#define CTILE 64

#if __has_builtin(__builtin_amdgcn_global_load_async_to_lds_b128) && \
    __has_builtin(__builtin_amdgcn_s_wait_asynccnt)
#define USE_ASYNC 1
#else
#define USE_ASYNC 0
#endif

__device__ __forceinline__ void cp16(const _Float16* g, _Float16* l) {
#if USE_ASYNC
  __builtin_amdgcn_global_load_async_to_lds_b128(
      (__attribute__((address_space(1))) i4*)g,
      (__attribute__((address_space(3))) i4*)l, 0, 0);
#else
  *(f4*)l = *(const f4*)g;
#endif
}

// ------- Phase 1: bin_active + NCHW -> padded NHWC f16 [32][58][58][256] ----
__global__ __launch_bounds__(256) void binact_kernel(const float* __restrict__ x,
                                                     _Float16* __restrict__ xbp) {
  __shared__ _Float16 tile[WW * 258];          // [w][c] padded rows (banks)
  const int nhp = blockIdx.x;                  // 0 .. 32*58-1
  const int n = nhp / HP, hp = nhp % HP;
  const int t = threadIdx.x;
  _Float16* dst = xbp + (size_t)nhp * (HP * CIN);
  if (hp == 0 || hp == HP - 1) {               // top/bottom halo rows
    for (int o = t; o < HP * CIN; o += 256) dst[o] = (_Float16)0.f;
    return;
  }
  const int h = hp - 1;
  for (int idx = t; idx < CIN * WW; idx += 256) {
    int c = idx / WW, w = idx - c * WW;
    const float* row = x + ((size_t)(n * CIN + c) * HH + h) * WW;
    float v = row[w];
    float p = row[w ^ 1];                      // pair partner along W (BIN_AGG=2)
    float m = 0.5f * (fabsf(v) + fabsf(p));
    float r = (v > 0.f) ? m : ((v < 0.f) ? -m : 0.f);   // sign(0)=0
    tile[w * 258 + c] = (_Float16)r;
  }
  __syncthreads();
  for (int o = t; o < HP * CIN; o += 256) {    // wp=0 and wp=57 are zero halo
    int wp = o >> 8, c = o & 255;
    dst[o] = (wp == 0 || wp == HP - 1) ? (_Float16)0.f : tile[(wp - 1) * 258 + c];
  }
}

// ------- Phase 2: weights OIHW f32 -> [tap*8+chunk][cout][k32] f16 ----------
__global__ __launch_bounds__(256) void wprep_kernel(const float* __restrict__ wsrc,
                                                    _Float16* __restrict__ wt) {
  int idx = blockIdx.x * 256 + threadIdx.x;    // 9*8*256*32 = 589824 total
  if (idx >= 9 * 8 * 256 * 32) return;
  int kk    = idx & 31;
  int co    = (idx >> 5) & 255;
  int chunk = (idx >> 13) & 7;
  int tap   = idx >> 16;
  int ci    = chunk * 32 + kk;
  wt[idx] = (_Float16)wsrc[((size_t)co * CIN + ci) * 9 + tap];
}

// -------- Phase 3: implicit GEMM conv, async double-buffered WMMA -----------
// D[co][spatial] : A = weights (M=co x K), B = activations (K x N=spatial)
// Workgroup tile: 64 cout x 256 spatial; per-wave 32x64 = 2x4 WMMA accums.
__global__ __launch_bounds__(256) void conv_wmma_kernel(
    const _Float16* __restrict__ xbp, const _Float16* __restrict__ wt,
    const float* __restrict__ bias, float* __restrict__ out) {
  // padded LDS rows: 40 halves (80 B) per 32-half row -> conflict-free b128 reads
  __shared__ __align__(16) _Float16 ldsA[2][CTILE * 40];   // weights [co][k]
  __shared__ __align__(16) _Float16 ldsB[2][STILE * 40];   // acts    [s][k]

  const int bid    = blockIdx.x;
  const int stileI = bid >> 2;                 // 392 spatial tiles
  const int ctileI = bid & 3;                  // 4 cout tiles
  const int sbase  = stileI * STILE;
  const int cobase = ctileI * CTILE;

  const int t       = threadIdx.x;
  const int lane    = t & 31;
  const int wid     = t >> 5;
  const int waveC   = wid & 1;                 // cout sub-tile (2)
  const int waveS   = wid >> 1;                // spatial sub-tile (4) of 64
  const int halfsel = lane >> 4;
  const int l16     = lane & 15;
  const int ho      = halfsel * 8;             // fragment half-offset (halves)

  // ---- per-thread copy-slot geometry (fixed across the whole k-loop) ----
  const int row  = t >> 2;                     // 0..63
  const int part = t & 3;                      // 16B segment within 64B row
  const int ldsAoff = row * 40 + part * 8;

  // activation global bases at (kh=0,kw=0) for my four rows of the 256-row tile
  size_t baseS[4];
#pragma unroll
  for (int j = 0; j < 4; ++j) {
    int s = sbase + row + j * 64;
    int n = s / HW; int r = s - n * HW; int h = r / WW; int w = r - h * WW;
    baseS[j] = ((size_t)(n * HP + h) * HP + w) * CIN + part * 8;
  }
  const _Float16* wseg = wt + (size_t)cobase * 32 + t * 8;   // += 8192/step

  v8f acc[2][4] = {};                           // [mt(co16)][nt(spatial16)]

  // ---- prologue: stage k-step 0 into buffer 0 (tap 0 => off 0) ----
  cp16(wseg, &ldsA[0][ldsAoff]);
#pragma unroll
  for (int j = 0; j < 4; ++j)
    cp16(xbp + baseS[j], &ldsB[0][(row + j * 64) * 40 + part * 8]);

#pragma unroll 2
  for (int kk = 0; kk < 72; ++kk) {            // 9 taps * 8 ci-chunks, K=32
    const int p = kk & 1;
    if (kk + 1 < 72) {                         // stage next step into other buf
      const int k2 = kk + 1;
      const int tap = k2 >> 3, chunk = k2 & 7;
      const int off = ((tap / 3) * HP + (tap % 3)) * CIN + chunk * 32;  // uniform
      const int q = p ^ 1;
      cp16(wseg + (size_t)k2 * 8192, &ldsA[q][ldsAoff]);
#pragma unroll
      for (int j = 0; j < 4; ++j)
        cp16(xbp + baseS[j] + off, &ldsB[q][(row + j * 64) * 40 + part * 8]);
    }
#if USE_ASYNC
    if (kk + 1 < 72) __builtin_amdgcn_s_wait_asynccnt(5);   // drain stage kk only
    else             __builtin_amdgcn_s_wait_asynccnt(0);
#endif
    __syncthreads();                           // stage kk visible to all waves

    // fragment loads (ISA 16-bit layout: row = lane%16; halves 0-7 -> K
    // {0..7|8..15}, halves 8-15 -> K {16..23|24..31} by lane half)
    v16h afrag[2], bfrag[4];
#pragma unroll
    for (int mt = 0; mt < 2; ++mt) {
      const _Float16* rowp = &ldsA[p][(waveC * 32 + mt * 16 + l16) * 40];
      ((h8*)&afrag[mt])[0] = *(const h8*)&rowp[ho];
      ((h8*)&afrag[mt])[1] = *(const h8*)&rowp[ho + 16];
    }
#pragma unroll
    for (int nt = 0; nt < 4; ++nt) {
      const _Float16* rowp = &ldsB[p][(waveS * 64 + nt * 16 + l16) * 40];
      ((h8*)&bfrag[nt])[0] = *(const h8*)&rowp[ho];
      ((h8*)&bfrag[nt])[1] = *(const h8*)&rowp[ho + 16];
    }
#pragma unroll
    for (int mt = 0; mt < 2; ++mt)
#pragma unroll
      for (int nt = 0; nt < 4; ++nt)
        acc[mt][nt] = __builtin_amdgcn_wmma_f32_16x16x32_f16(
            false, afrag[mt], false, bfrag[nt], (short)0, acc[mt][nt],
            false, false);
    __syncthreads();                           // all reads of buf p done
  }

  // ---- epilogue: D lanes (N = lane%16) -> consecutive w, coalesced NCHW ----
#pragma unroll
  for (int mt = 0; mt < 2; ++mt) {
    int co = cobase + waveC * 32 + mt * 16 + halfsel * 8;    // + v below
    float bv[8];
#pragma unroll
    for (int v = 0; v < 8; ++v) bv[v] = bias[co + v];
#pragma unroll
    for (int nt = 0; nt < 4; ++nt) {
      int s = sbase + waveS * 64 + nt * 16 + l16;
      int n = s / HW; int r = s - n * HW;                    // r = h*56 + w
#pragma unroll
      for (int v = 0; v < 8; ++v)
        out[((size_t)(n * COUT) + co + v) * HW + r] = acc[mt][nt][v] + bv[v];
    }
  }
}

extern "C" void kernel_launch(void* const* d_in, const int* in_sizes, int n_in,
                              void* d_out, int out_size, void* d_ws, size_t ws_size,
                              hipStream_t stream) {
  const float* x    = (const float*)d_in[0];
  const float* wgt  = (const float*)d_in[1];
  const float* bias = (const float*)d_in[2];
  float* out        = (float*)d_out;

  const size_t xbp_bytes = (size_t)NB * HP * HP * CIN * 2;   // 55,115,776 B
  _Float16* xbp = (_Float16*)d_ws;
  _Float16* wt  = (_Float16*)((char*)d_ws + xbp_bytes);      // +1,179,648 B

  binact_kernel<<<NB * HP, 256, 0, stream>>>(x, xbp);
  wprep_kernel<<<(9 * 8 * 256 * 32) / 256, 256, 0, stream>>>(wgt, wt);
  conv_wmma_kernel<<<(SPATIAL / STILE) * (COUT / CTILE), 256, 0, stream>>>(
      xbp, wt, bias, out);
}
